// MALA_40467181862986
// MI455X (gfx1250) — matrix operand order repaired and checked by reading, not compiled, not run on hardware
//
#include <hip/hip_runtime.h>
#include <hip/hip_bf16.h>
#include <math.h>

// ---------------------------------------------------------------------------
// MALA linear-attention block for MI455X (gfx1250), wave32 + WMMA f16 (f32 acc).
// Fragment layouts per CDNA5 ISA 7.12.2 (wave32):
//   A (16x32 f16):  lane: m=lane&15, hi=lane>>4; VGPR v: K = (v>>2)*16 + hi*8 + (v&3)*2 (+0,+1)
//   B (32x16 f16):  lane: n=lane&15, hi=lane>>4; VGPR v: K = (v>>2)*16 + hi*8 + (v&3)*2 (+0,+1)
//   C/D (16x16 f32): lane: n=lane&15, hi=lane>>4; VGPR r: m = r + 8*hi
// LDS tiles are K-major with row pitch 36 halfwords (8B-aligned float4 groups,
// even pitch keeps 32-bit pair loads 4B-aligned).
// ---------------------------------------------------------------------------

typedef __attribute__((ext_vector_type(16))) _Float16 v16h;
typedef __attribute__((ext_vector_type(4)))  _Float16 v4h;
typedef __attribute__((ext_vector_type(8)))  float    v8f;

union AFrag { v16h h; unsigned int u[8]; };
union Acc   { v8f  f; float e[8]; };

#define LDIM 4096      // H*W
#define KDIM 256       // input channels of both 1x1 convs
#define NBATCH 8
#define NHEAD 8
#define PITCH 36       // LDS row pitch in halfwords

static __device__ __forceinline__ float elu1(float v) {
  return v > 0.f ? v + 1.f : __expf(v);   // jax.nn.elu(v)+1
}

static __device__ __forceinline__ v4h cvt4(float4 f, float scl) {
  v4h h;
  h.x = (_Float16)(f.x * scl); h.y = (_Float16)(f.y * scl);
  h.z = (_Float16)(f.z * scl); h.w = (_Float16)(f.w * scl);
  return h;
}

// theta_shift on 4 consecutive l positions for head-dim j.
// rot[2i]=-x[2i+1], rot[2i+1]=x[2i]; dims 0..15 keyed on h=l>>6, 16..31 on w=l&63.
static __device__ __forceinline__ v4h rope4(float4 x, float4 p, int l, int j, float scl) {
  const float sgn  = (j & 1) ? 1.f : -1.f;
  const float freq = __expf(-(float)((j & 15) >> 1) * (9.210340371976184f / 7.0f)); // ln(1e4)/7
  float xs[4] = {x.x, x.y, x.z, x.w};
  float ps[4] = {p.x, p.y, p.z, p.w};
  v4h o;
#pragma unroll
  for (int t = 0; t < 4; ++t) {
    int lt  = l + t;
    int pos = (j < 16) ? (lt >> 6) : (lt & 63);
    float a = (float)pos * freq;
    o[t] = (_Float16)((xs[t] * __cosf(a) + sgn * ps[t] * __sinf(a)) * scl);
  }
  return o;
}

static __device__ __forceinline__ void loadAfrag(AFrag &a, const _Float16* base,
                                                 int row, int hi) {
#pragma unroll
  for (int v = 0; v < 8; ++v)
    a.u[v] = *(const unsigned int*)&base[row * PITCH + (((v >> 2) << 4) + ((v & 3) << 1) + hi * 8)];
}
static __device__ __forceinline__ void loadBfrag(AFrag &b, const _Float16* base,
                                                 int row, int hi) {
#pragma unroll
  for (int v = 0; v < 8; ++v)
    b.u[v] = *(const unsigned int*)&base[row * PITCH + (((v >> 2) << 4) + ((v & 3) << 1) + hi * 8)];
}

// ---------------------------------------------------------------------------
// 1x1 conv as GEMM: Y[b,o,l] = sum_c W[o,c] X[b,c,l] + bias[o]; elu+1 for
// o < elu_limit. Block 256 thr (8 waves), tile M=64 x N=32, K=256 in 8 steps,
// double-buffered LDS + register prefetch (one barrier per K-step).
// ---------------------------------------------------------------------------
__global__ __launch_bounds__(256)
void conv1x1_wmma(const float* __restrict__ W, const float* __restrict__ bias,
                  const float* __restrict__ X, float* __restrict__ Y,
                  int M, int elu_limit)
{
  __shared__ __align__(16) _Float16 As[2][64 * PITCH];
  __shared__ __align__(16) _Float16 Bs[2][32 * PITCH];

  const int tid  = threadIdx.x;
  const int lane = tid & 31;
  const int wv   = tid >> 5;           // wave32
  const int hi   = lane >> 4;
  const int ln   = lane & 15;
  const int wm   = wv >> 1;            // 0..3 M subtile
  const int wn   = wv & 1;             // 0..1 N subtile
  const int l0   = blockIdx.x * 32;
  const int o0   = blockIdx.y * 64;
  const int b    = blockIdx.z;
  const float* Xb = X + (size_t)b * KDIM * LDIM;

  // A fill: float4 id f in [0,512): m=f>>3, kq=(f&7)*4  (rows contiguous in k)
  const int ma0 = tid >> 3,         ka0 = (tid & 7) * 4;
  const int ma1 = (tid + 256) >> 3, ka1 = ka0;           // f1 = tid+256
  // B fill: float4 id g=tid in [0,256): k=g>>3, lq=(g&7)*4 (contiguous in l)
  const int kb0 = tid >> 3, lq0 = (tid & 7) * 4;

  const float* Wa0 = &W[(size_t)(o0 + ma0) * KDIM + ka0];
  const float* Wa1 = &W[(size_t)(o0 + ma1) * KDIM + ka1];
  const float* Xg  = &Xb[(size_t)kb0 * LDIM + l0 + lq0];

  float4 ra0 = *(const float4*)Wa0;
  float4 ra1 = *(const float4*)Wa1;
  float4 rb  = *(const float4*)Xg;

  Acc acc;
#pragma unroll
  for (int r = 0; r < 8; ++r) acc.e[r] = 0.f;

#pragma unroll
  for (int it = 0; it < KDIM / 32; ++it) {
    const int cur = it & 1;
    *(v4h*)&As[cur][ma0 * PITCH + ka0] = cvt4(ra0, 1.f);
    *(v4h*)&As[cur][ma1 * PITCH + ka1] = cvt4(ra1, 1.f);
    {
      v4h hb = cvt4(rb, 1.f);
      Bs[cur][(lq0 + 0) * PITCH + kb0] = hb.x;
      Bs[cur][(lq0 + 1) * PITCH + kb0] = hb.y;
      Bs[cur][(lq0 + 2) * PITCH + kb0] = hb.z;
      Bs[cur][(lq0 + 3) * PITCH + kb0] = hb.w;
    }
    __syncthreads();
    if (it + 1 < KDIM / 32) {               // prefetch next K-tile into regs
      ra0 = *(const float4*)(Wa0 + (size_t)(it + 1) * 32);
      ra1 = *(const float4*)(Wa1 + (size_t)(it + 1) * 32);
      rb  = *(const float4*)(Xg + (size_t)(it + 1) * 32 * LDIM);
    }
    AFrag a, bv;
    loadAfrag(a,  As[cur], wm * 16 + ln, hi);
    loadBfrag(bv, Bs[cur], wn * 16 + ln, hi);
    acc.f = __builtin_amdgcn_wmma_f32_16x16x32_f16(false, a.h, false, bv.h,
                                                   (short)0, acc.f, false, false);
    // double buffering: one barrier per step is sufficient
  }

#pragma unroll
  for (int r = 0; r < 8; ++r) {
    int o = o0 + wm * 16 + r + 8 * hi;
    int l = l0 + wn * 16 + ln;
    float y = acc.e[r] + bias[o];
    if (o < elu_limit) y = elu1(y);
    Y[((size_t)b * M + o) * LDIM + l] = y;   // coalesced across lanes
  }
}

// ---------------------------------------------------------------------------
// Per-(b,channel) mean over L for k and v channels: mean[b][c], c in [0,512),
// source = qkvo channel 256+c. float4 loads.
// ---------------------------------------------------------------------------
__global__ __launch_bounds__(256)
void mean_kernel(const float* __restrict__ qkvo, float* __restrict__ mean)
{
  __shared__ float red[256];
  int c = blockIdx.x & 511;
  int b = blockIdx.x >> 9;
  const float* p = qkvo + ((size_t)b * 1024 + 256 + c) * LDIM;
  float s = 0.f;
  for (int l = threadIdx.x * 4; l < LDIM; l += 1024) {
    float4 v = *(const float4*)&p[l];
    s += (v.x + v.y) + (v.z + v.w);
  }
  red[threadIdx.x] = s;
  __syncthreads();
  for (int st = 128; st > 0; st >>= 1) {
    if (threadIdx.x < st) red[threadIdx.x] += red[threadIdx.x + st];
    __syncthreads();
  }
  if (threadIdx.x == 0) mean[(size_t)b * 512 + c] = red[0] * (1.f / (float)LDIM);
}

// ---------------------------------------------------------------------------
// Depthwise 5x5 (pad 2) on v channels -> lepe[b][c][l]
// ---------------------------------------------------------------------------
__global__ __launch_bounds__(256)
void lepe_kernel(const float* __restrict__ qkvo, const float* __restrict__ lw,
                 const float* __restrict__ lb, float* __restrict__ lepe)
{
  __shared__ float wsm[25];
  int c = blockIdx.y;
  int b = blockIdx.z;
  if (threadIdx.x < 25) wsm[threadIdx.x] = lw[c * 25 + threadIdx.x];
  __syncthreads();

  int l = blockIdx.x * 256 + threadIdx.x;
  int h = l >> 6, w = l & 63;
  const float* vch = qkvo + ((size_t)b * 1024 + 512 + c) * LDIM;
  float s = lb[c];
#pragma unroll
  for (int dy = 0; dy < 5; ++dy) {
    int hy = h + dy - 2;
    if (hy < 0 || hy >= 64) continue;
#pragma unroll
    for (int dx = 0; dx < 5; ++dx) {
      int wx = w + dx - 2;
      if (wx < 0 || wx >= 64) continue;
      s += wsm[dy * 5 + dx] * vch[hy * 64 + wx];
    }
  }
  lepe[((size_t)b * 256 + c) * LDIM + l] = s;
}

// ---------------------------------------------------------------------------
// kv[b,n,d,e] = sum_l (rope(k)[l,d]*cf)*(v[l,e]*cf), K=4096 in 128 steps.
// One block (4 waves) per (b,head); 2x2 WMMA subtiles; double-buffered LDS
// with register prefetch, one barrier per step.
// ---------------------------------------------------------------------------
__global__ __launch_bounds__(128)
void kv_kernel(const float* __restrict__ qkvo, float* __restrict__ kv)
{
  __shared__ __align__(16) _Float16 Ak[2][32 * PITCH];
  __shared__ __align__(16) _Float16 Bv[2][32 * PITCH];

  const int tid  = threadIdx.x;
  const int lane = tid & 31;
  const int wv   = tid >> 5;
  const int hi   = lane >> 4;
  const int ln   = lane & 15;
  const int wm   = wv >> 1, wn = wv & 1;
  const int n = blockIdx.x & 7, b = blockIdx.x >> 3;

  const float* kb_ = qkvo + ((size_t)b * 1024 + 256 + n * 32) * LDIM;
  const float* vb_ = qkvo + ((size_t)b * 1024 + 512 + n * 32) * LDIM;
  const float scale = 0.17677669529663687f;     // 32^-0.5
  const float cf = sqrtf(scale / (float)LDIM);  // (scale/L)^0.5

  // Tile fill: float4 id f = tid + j*128, f in [0,256): d=f>>3, kq=(f&7)*4
  int dA[2], kqA[2];
#pragma unroll
  for (int j = 0; j < 2; ++j) { int f = tid + j * 128; dA[j] = f >> 3; kqA[j] = (f & 7) * 4; }

  float4 kx[2], kp[2], vx[2];
#pragma unroll
  for (int j = 0; j < 2; ++j) {
    kx[j] = *(const float4*)&kb_[(size_t)dA[j] * LDIM + kqA[j]];
    kp[j] = *(const float4*)&kb_[(size_t)(dA[j] ^ 1) * LDIM + kqA[j]];
    vx[j] = *(const float4*)&vb_[(size_t)dA[j] * LDIM + kqA[j]];
  }

  Acc acc;
#pragma unroll
  for (int r = 0; r < 8; ++r) acc.e[r] = 0.f;

  for (int l0 = 0; l0 < LDIM; l0 += 32) {
    const int cur = (l0 >> 5) & 1;
#pragma unroll
    for (int j = 0; j < 2; ++j) {
      *(v4h*)&Ak[cur][dA[j] * PITCH + kqA[j]] = rope4(kx[j], kp[j], l0 + kqA[j], dA[j], cf);
      *(v4h*)&Bv[cur][dA[j] * PITCH + kqA[j]] = cvt4(vx[j], cf);
    }
    __syncthreads();
    if (l0 + 32 < LDIM) {
#pragma unroll
      for (int j = 0; j < 2; ++j) {
        kx[j] = *(const float4*)&kb_[(size_t)dA[j] * LDIM + l0 + 32 + kqA[j]];
        kp[j] = *(const float4*)&kb_[(size_t)(dA[j] ^ 1) * LDIM + l0 + 32 + kqA[j]];
        vx[j] = *(const float4*)&vb_[(size_t)dA[j] * LDIM + l0 + 32 + kqA[j]];
      }
    }
    AFrag a, bvf;
    loadAfrag(a,   Ak[cur], wm * 16 + ln, hi);
    loadBfrag(bvf, Bv[cur], wn * 16 + ln, hi);
    acc.f = __builtin_amdgcn_wmma_f32_16x16x32_f16(false, a.h, false, bvf.h,
                                                   (short)0, acc.f, false, false);
  }

#pragma unroll
  for (int r = 0; r < 8; ++r) {
    int d = wm * 16 + r + 8 * hi;
    int e = wn * 16 + ln;
    kv[(((size_t)b * NHEAD + n) * 32 + d) * 32 + e] = acc.e[r];
  }
}

// ---------------------------------------------------------------------------
// z[b,n,l] = scale * sum_d q_pre_rope[l,d] * kmean[b,n,d]  (float4 over l)
// ---------------------------------------------------------------------------
__global__ __launch_bounds__(256)
void z_kernel(const float* __restrict__ qkvo, const float* __restrict__ mean,
              float* __restrict__ z)
{
  int n = blockIdx.y, b = blockIdx.z;
  int l = (blockIdx.x * 256 + threadIdx.x) * 4;
  const float* qb_ = qkvo + ((size_t)b * 1024 + n * 32) * LDIM;
  const float* km  = mean + (size_t)b * 512 + n * 32;    // kmean slice
  float ax = 0.f, ay = 0.f, az = 0.f, aw = 0.f;
#pragma unroll
  for (int d = 0; d < 32; ++d) {
    float4 xq = *(const float4*)&qb_[(size_t)d * LDIM + l];
    float m = km[d];
    ax += xq.x * m; ay += xq.y * m; az += xq.z * m; aw += xq.w * m;
  }
  const float scale = 0.17677669529663687f;
  float4 o; o.x = ax * scale; o.y = ay * scale; o.z = az * scale; o.w = aw * scale;
  *(float4*)&z[((size_t)b * NHEAD + n) * LDIM + l] = o;
}

// ---------------------------------------------------------------------------
// u = ((rope(q) @ kv) * (1 + 1/(z+1e-6)) - z*vmean + lepe) * o_gate
// Block 256 thr (8 waves): 64 l-rows x 32 e-cols, one WMMA per wave (K=32).
// Epilogue: per-lane rows r=0..7 are 8 consecutive l -> two float4 ld/st.
// ---------------------------------------------------------------------------
__global__ __launch_bounds__(256)
void out_kernel(const float* __restrict__ qkvo, const float* __restrict__ kv,
                const float* __restrict__ z, const float* __restrict__ mean,
                const float* __restrict__ lepe, float* __restrict__ u)
{
  __shared__ __align__(16) _Float16 Aq[64 * PITCH];
  __shared__ __align__(16) _Float16 Kt[32 * PITCH];

  const int tid  = threadIdx.x;
  const int lane = tid & 31;
  const int wv   = tid >> 5;
  const int hi   = lane >> 4;
  const int ln   = lane & 15;
  const int lt   = wv >> 1;          // 0..3 l-subtile
  const int eh   = wv & 1;           // 0..1 e-half
  const int l0 = blockIdx.x * 64;
  const int n  = blockIdx.y, b = blockIdx.z;

  const float* qb_ = qkvo + ((size_t)b * 1024 + n * 32) * LDIM;
  const float* kvb = kv + (((size_t)b * NHEAD + n) * 32) * 32;

  // Aq fill: f = tid + j*256 in [0,512): d=f>>4, lc=f&15, l=l0+lc*4.
  // Load float4 along l (contiguous), rope, 4 transposed b16 stores.
#pragma unroll
  for (int j = 0; j < 2; ++j) {
    int f = tid + j * 256;
    int d = f >> 4, lc = f & 15;
    int l = l0 + lc * 4;
    float4 x = *(const float4*)&qb_[(size_t)d * LDIM + l];
    float4 p = *(const float4*)&qb_[(size_t)(d ^ 1) * LDIM + l];
    v4h h = rope4(x, p, l, d, 1.f);
    Aq[(lc * 4 + 0) * PITCH + d] = h.x;
    Aq[(lc * 4 + 1) * PITCH + d] = h.y;
    Aq[(lc * 4 + 2) * PITCH + d] = h.z;
    Aq[(lc * 4 + 3) * PITCH + d] = h.w;
  }
  // Kt fill (B^T: [e][d]): f = tid in [0,256): d=f>>3, eq=(f&7)*4
  {
    int d = tid >> 3, eq = (tid & 7) * 4;
    float4 x = *(const float4*)&kvb[d * 32 + eq];
    v4h h = cvt4(x, 1.f);
    Kt[(eq + 0) * PITCH + d] = h.x;
    Kt[(eq + 1) * PITCH + d] = h.y;
    Kt[(eq + 2) * PITCH + d] = h.z;
    Kt[(eq + 3) * PITCH + d] = h.w;
  }
  __syncthreads();

  Acc acc;
#pragma unroll
  for (int r = 0; r < 8; ++r) acc.e[r] = 0.f;

  AFrag a, bvf;
  loadAfrag(a,   Aq, lt * 16 + ln, hi);
  loadBfrag(bvf, Kt, eh * 16 + ln, hi);
  acc.f = __builtin_amdgcn_wmma_f32_16x16x32_f16(false, a.h, false, bvf.h,
                                                 (short)0, acc.f, false, false);

  // Epilogue: lane owns e = eh*16+ln and 8 consecutive l = lb..lb+7
  const int e  = eh * 16 + ln;
  const int c  = n * 32 + e;
  const int lb = l0 + lt * 16 + 8 * hi;
  const float vm = mean[(size_t)b * 512 + 256 + c];          // vmean
  const float* zp = &z[((size_t)b * NHEAD + n) * LDIM + lb];
  const float* lp = &lepe[((size_t)b * 256 + c) * LDIM + lb];
  const float* op = &qkvo[((size_t)b * 1024 + 768 + c) * LDIM + lb];
  float* up = &u[((size_t)b * 256 + c) * LDIM + lb];

#pragma unroll
  for (int g = 0; g < 2; ++g) {
    float4 zv = *(const float4*)&zp[g * 4];
    float4 le = *(const float4*)&lp[g * 4];
    float4 og = *(const float4*)&op[g * 4];
    float res[4] = {acc.e[g * 4 + 0], acc.e[g * 4 + 1], acc.e[g * 4 + 2], acc.e[g * 4 + 3]};
    float zz[4] = {zv.x, zv.y, zv.z, zv.w};
    float ll[4] = {le.x, le.y, le.z, le.w};
    float oo[4] = {og.x, og.y, og.z, og.w};
    float4 out;
    float* po = &out.x;
#pragma unroll
    for (int t = 0; t < 4; ++t) {
      float r2 = res[t] * (1.f + 1.f / (zz[t] + 1e-6f)) - zz[t] * vm;
      po[t] = (r2 + ll[t]) * oo[t];
    }
    *(float4*)&up[g * 4] = out;
  }
}

// ---------------------------------------------------------------------------
// Launch: qkvo GEMM -> means -> lepe -> kv -> z -> fused output -> proj GEMM
// Workspace (floats): qkvo 32M | mean 4K | lepe 8M | z 256K | kv 64K | u 8M
// ---------------------------------------------------------------------------
extern "C" void kernel_launch(void* const* d_in, const int* in_sizes, int n_in,
                              void* d_out, int out_size, void* d_ws, size_t ws_size,
                              hipStream_t stream)
{
  (void)in_sizes; (void)n_in; (void)out_size; (void)ws_size;

  const float* x      = (const float*)d_in[0];
  const float* qkvo_w = (const float*)d_in[1];
  const float* qkvo_b = (const float*)d_in[2];
  const float* lepe_w = (const float*)d_in[3];
  const float* lepe_b = (const float*)d_in[4];
  const float* proj_w = (const float*)d_in[5];
  const float* proj_b = (const float*)d_in[6];
  float* out = (float*)d_out;

  float* ws = (float*)d_ws;
  size_t off = 0;
  float* qkvo = ws + off; off += (size_t)NBATCH * 1024 * LDIM;
  float* mean = ws + off; off += (size_t)NBATCH * 512;
  float* lepe = ws + off; off += (size_t)NBATCH * 256 * LDIM;
  float* zbuf = ws + off; off += (size_t)NBATCH * NHEAD * LDIM;
  float* kvb  = ws + off; off += (size_t)NBATCH * NHEAD * 32 * 32;
  float* ubuf = ws + off; off += (size_t)NBATCH * 256 * LDIM;

  // 1) qkvo = W4 @ x + b4, elu+1 fused on q,k channels (o < 512)
  conv1x1_wmma<<<dim3(LDIM / 32, 1024 / 64, NBATCH), 256, 0, stream>>>(
      qkvo_w, qkvo_b, x, qkvo, 1024, 512);

  // 2) kmean / vmean
  mean_kernel<<<dim3(NBATCH * 512), 256, 0, stream>>>(qkvo, mean);

  // 3) lepe = depthwise5x5(v)
  lepe_kernel<<<dim3(LDIM / 256, 256, NBATCH), 256, 0, stream>>>(
      qkvo, lepe_w, lepe_b, lepe);

  // 4) kv = (rope(k)*c)^T (v*c)
  kv_kernel<<<dim3(NBATCH * NHEAD), 128, 0, stream>>>(qkvo, kvb);

  // 5) z = scale * q_pre . kmean
  z_kernel<<<dim3(LDIM / 1024, NHEAD, NBATCH), 256, 0, stream>>>(qkvo, mean, zbuf);

  // 6) u = ((rope(q)@kv)*(1+1/(z+eps)) - z*vmean + lepe) * o
  out_kernel<<<dim3(LDIM / 64, NHEAD, NBATCH), 256, 0, stream>>>(
      qkvo, kvb, zbuf, mean, lepe, ubuf);

  // 7) out = proj_w @ u + proj_b
  conv1x1_wmma<<<dim3(LDIM / 32, 256 / 64, NBATCH), 256, 0, stream>>>(
      proj_w, proj_b, ubuf, out, 256, 0);
}
